// MambaAdaptiveScanpathGenerator_79783312491125
// MI455X (gfx1250) — compile-verified, matrix operand order checked by reading, not run on hardware
//
#include <hip/hip_runtime.h>

// ---------------- model constants ----------------
constexpr int Dm  = 384;
constexpr int DSm = 256;
constexpr int DIm = 768;
constexpr int DTR = 24;
constexpr int PS  = 224;
constexpr int IMH = 256;
constexpr int IMW = 512;
constexpr int FS  = 14;
constexpr int NHm = 8;
constexpr int HDm = 48;
constexpr int Bm_ = 16;
constexpr int Nm_ = FS * FS;   // 196
constexpr int Tm_ = 8;

typedef __attribute__((ext_vector_type(16))) __bf16 v16bf;
typedef __attribute__((ext_vector_type(8)))  float  v8f;
typedef __attribute__((ext_vector_type(8)))  unsigned int v8u;
typedef __attribute__((ext_vector_type(4)))  unsigned int v4u;

// ---------------- small device helpers ----------------
__device__ inline unsigned short f2bf(float x) {
  union { float f; unsigned int u; } c; c.f = x;
  unsigned int u = c.u;
  unsigned int r = (u + 0x7FFFu + ((u >> 16) & 1u)) >> 16;  // round-nearest-even
  return (unsigned short)r;
}

__device__ inline float geluf(float x) { return 0.5f * x * (1.f + erff(x * 0.70710678f)); }
__device__ inline float sigf(float x)  { return 1.f / (1.f + __expf(-x)); }

__device__ inline unsigned int hash3(unsigned int a, unsigned int b, unsigned int c) {
  unsigned int h = a * 0x9E3779B1u ^ b * 0x85EBCA77u ^ c * 0xC2B2AE3Du;
  h ^= h >> 16; h *= 0x7FEB352Du; h ^= h >> 15; h *= 0x846CA68Bu; h ^= h >> 16;
  return h;
}
__device__ inline float u01(unsigned int h) { return (float)(h >> 8) * (1.0f / 16777216.0f); }
__device__ inline float nrmf(unsigned int a, unsigned int b, unsigned int c) {
  float u1 = fmaxf(u01(hash3(a, b, c)), 1e-7f);
  float u2 = u01(hash3(a ^ 0xDEADBEEFu, b, c));
  return sqrtf(-2.f * logf(u1)) * __cosf(6.2831853f * u2);
}

template<int NW>
__device__ inline float blockSum(float v, float* red) {
  #pragma unroll
  for (int o = 16; o > 0; o >>= 1) v += __shfl_xor(v, o, 32);
  int w = threadIdx.x >> 5;
  if ((threadIdx.x & 31) == 0) red[w] = v;
  __syncthreads();
  float r = 0.f;
  #pragma unroll
  for (int i = 0; i < NW; i++) r += red[i];
  __syncthreads();
  return r;
}
template<int NW>
__device__ inline float blockMax(float v, float* red) {
  #pragma unroll
  for (int o = 16; o > 0; o >>= 1) v = fmaxf(v, __shfl_xor(v, o, 32));
  int w = threadIdx.x >> 5;
  if ((threadIdx.x & 31) == 0) red[w] = v;
  __syncthreads();
  float r = -3.4e38f;
  #pragma unroll
  for (int i = 0; i < NW; i++) r = fmaxf(r, red[i]);
  __syncthreads();
  return r;
}

__device__ inline void affineParams(const float* prev, int b,
                                    float& t00, float& t11, float& t02, float& t12) {
  float pc0 = prev[b * 2 + 0] * (float)IMH - 0.5f * (float)PS;
  float pc1 = prev[b * 2 + 1] * (float)IMW - 0.5f * (float)PS;
  float xc = fminf(fmaxf(pc0, 0.f), (float)(IMH - PS));
  float yc = fminf(fmaxf(pc1, 0.f), (float)(IMW - PS));
  t00 = (float)PS / (float)IMW;
  t11 = (float)PS / (float)IMH;
  t02 = -1.f + (2.f * yc + (float)PS) / (float)IMW;
  t12 = -1.f + (2.f * xc + (float)PS) / (float)IMH;
}

__device__ inline float bilin(const float* img, int H, int W, float iy, float ix) {
  float fx = floorf(ix), fy = floorf(iy);
  int x0 = (int)fx, y0 = (int)fy;
  float wx1 = ix - fx, wy1 = iy - fy;
  float v = 0.f;
  #pragma unroll
  for (int dy = 0; dy < 2; dy++)
    #pragma unroll
    for (int dx = 0; dx < 2; dx++) {
      int xx = x0 + dx, yy = y0 + dy;
      if (xx >= 0 && xx < W && yy >= 0 && yy < H) {
        float w = (dx ? wx1 : 1.f - wx1) * (dy ? wy1 : 1.f - wy1);
        v += img[yy * W + xx] * w;
      }
    }
  return v;
}

// ---------------- CDNA5 async memory->LDS helpers ----------------
__device__ inline unsigned ldsOffset(const void* p) {
  // addrspace(3) flat addresses carry the LDS byte offset in the low 32 bits
  return (unsigned)(unsigned long long)p;
}
__device__ inline void asyncLoadB128(unsigned lds_off, const void* gaddr) {
  asm volatile("global_load_async_to_lds_b128 %0, %1, off"
               :: "v"(lds_off), "v"(gaddr) : "memory");
}
__device__ inline void waitAsync3() { asm volatile("s_wait_asynccnt 0x3" ::: "memory"); }
__device__ inline void waitAsync0() { asm volatile("s_wait_asynccnt 0x0" ::: "memory"); }

// ---------------- generic bf16 WMMA GEMM ----------------
// C[M,N](f32) = act(A[M,K](bf16,row) * Bp + bias).  Bp is the weight pre-packed in
// per-lane WMMA-fragment order: blocks of (k-tile 32 x n-tile 16) = 32 lanes x 32B,
// block index = kt * (Npad/16) + n16, zero-padded to Npad = ceil(N/64)*64.
// Workgroup: 128 threads = 4 waves; C tile 32(M) x 64(N); 2 accumulators/wave;
// single async double-buffered pipeline (3 async b128 batches per wave per K-tile).
// Requires K % 32 == 0 (true for every GEMM in this model).
constexpr int AS_LDA = 40;  // ushorts, row = 80B (16B multiple)

template<int ACT>
__global__ __launch_bounds__(128)
void k_gemm(const unsigned short* __restrict__ A, int lda,
            const unsigned short* __restrict__ Bp,
            const float* __restrict__ bias,
            float* __restrict__ C, unsigned short* __restrict__ Cbf, int ldc,
            int M, int N, int K) {
  __shared__ __align__(16) unsigned short As[2][32][AS_LDA];
  __shared__ __align__(16) unsigned short Bs[2][2048];   // 4KB per stage
  const int tid  = threadIdx.x;
  const int wave = tid >> 5;
  const int lane = tid & 31;
  const int m0 = blockIdx.y * 32;
  const int n0 = blockIdx.x * 64;
  const int mloc = lane & 15;
  const int hi   = lane >> 4;
  const int nloc = (wave << 4) | mloc;
  const int NT   = K >> 5;
  const int NT16 = ((N + 63) >> 6) << 2;   // Npad/16

  v8f acc0 = {};
  v8f acc1 = {};

  // A: 32 rows x 4 chunks of 16B; one chunk per thread; clamp rows for M edges
  const int amm = tid >> 2, ac = tid & 3;
  int arow = m0 + amm; if (arow >= M) arow = M - 1;
  const unsigned short* agp = A + (size_t)arow * lda + ac * 8;
  const unsigned aoff = ldsOffset(&As[0][0][0]) + (unsigned)(amm * AS_LDA * 2 + ac * 16);
  const unsigned aBufStride = (unsigned)(32 * AS_LDA * 2);
  // B: 4KB per K-tile (4 consecutive n16 blocks), 256 chunks, 2 per thread
  const unsigned short* bbase = Bp + (size_t)(blockIdx.x * 4) * 512;
  const size_t bTileStride = (size_t)NT16 * 512;          // elements per k-tile
  const unsigned boff = ldsOffset(&Bs[0][0]);

  // prologue: tile 0 into buffer 0
  asyncLoadB128(aoff, agp);
  asyncLoadB128(boff + tid * 16u, bbase + (size_t)tid * 8);
  asyncLoadB128(boff + (tid + 128u) * 16u, bbase + (size_t)(tid + 128) * 8);

  for (int it = 0; it < NT; ++it) {
    const int buf = it & 1;
    if (it + 1 < NT) {
      const int nb = (it + 1) & 1;
      const size_t ks = (size_t)(it + 1) * 32;
      const unsigned short* bt = bbase + (size_t)(it + 1) * bTileStride;
      asyncLoadB128(aoff + nb * aBufStride, agp + ks);
      asyncLoadB128(boff + (unsigned)(nb * 4096) + tid * 16u, bt + (size_t)tid * 8);
      asyncLoadB128(boff + (unsigned)(nb * 4096) + (tid + 128u) * 16u,
                    bt + (size_t)(tid + 128) * 8);
      waitAsync3();
    } else {
      waitAsync0();
    }
    __syncthreads();  // tile `it` visible to all waves

    v8u au0, au1;
    #pragma unroll
    for (int v = 0; v < 8; ++v) {
      int kb = ((v < 4) ? (v * 2) : (16 + (v - 4) * 2)) + hi * 8;  // even
      au0[v] = *(const unsigned int*)&As[buf][mloc][kb];
      au1[v] = *(const unsigned int*)&As[buf][16 + mloc][kb];
    }
    const unsigned short* bp = &Bs[buf][(wave << 9) + (lane << 4)];  // lane's 32B
    v4u b0 = *(const v4u*)bp;
    v4u b1 = *(const v4u*)(bp + 8);
    v8u bu; bu[0] = b0[0]; bu[1] = b0[1]; bu[2] = b0[2]; bu[3] = b0[3];
            bu[4] = b1[0]; bu[5] = b1[1]; bu[6] = b1[2]; bu[7] = b1[3];
    v16bf bfr = __builtin_bit_cast(v16bf, bu);
    acc0 = __builtin_amdgcn_wmma_f32_16x16x32_bf16(false, __builtin_bit_cast(v16bf, au0),
                                                   false, bfr, (short)0, acc0, false, false);
    acc1 = __builtin_amdgcn_wmma_f32_16x16x32_bf16(false, __builtin_bit_cast(v16bf, au1),
                                                   false, bfr, (short)0, acc1, false, false);
    __syncthreads();  // all waves done reading buf before it is overwritten
  }

  int gn = n0 + nloc;
  if (gn < N) {
    float bv = bias ? bias[gn] : 0.f;
    #pragma unroll
    for (int r = 0; r < 8; ++r) {
      int gm = m0 + r + 8 * hi;
      if (gm < M) {
        float x = acc0[r] + bv;
        if (ACT == 1) x = geluf(x);
        else if (ACT == 2) x = sigf(x);
        C[(size_t)gm * ldc + gn] = x;
        if (Cbf) Cbf[(size_t)gm * ldc + gn] = f2bf(x);
      }
      int gm1 = m0 + 16 + r + 8 * hi;
      if (gm1 < M) {
        float x = acc1[r] + bv;
        if (ACT == 1) x = geluf(x);
        else if (ACT == 2) x = sigf(x);
        C[(size_t)gm1 * ldc + gn] = x;
        if (Cbf) Cbf[(size_t)gm1 * ldc + gn] = f2bf(x);
      }
    }
  }
}

// ---------------- weight re-pack: (dout,din) f32 -> WMMA-fragment bf16 ----------------
// dst element index: ((kt*(Npad/16) + n16)*32 + lane)*16 + h
//   n = n16*16 + (lane&15); hi = lane>>4; v = h>>1; p = h&1;
//   k = kt*32 + ((v<4)? 2v : 16+2(v-4)) + 8*hi + p
__global__ void k_wprep(const float* __restrict__ src, unsigned short* __restrict__ dst,
                        int N, int K, int Kpad, int Npad) {
  int idx = blockIdx.x * 256 + threadIdx.x;
  int total = Kpad * Npad;
  if (idx >= total) return;
  int h = idx & 15;
  int lane = (idx >> 4) & 31;
  int blk = idx >> 9;
  int nt16 = Npad >> 4;
  int n16 = blk % nt16, kt = blk / nt16;
  int n = n16 * 16 + (lane & 15);
  int hi = lane >> 4;
  int v = h >> 1, p = h & 1;
  int k = kt * 32 + ((v < 4) ? v * 2 : 16 + (v - 4) * 2) + hi * 8 + p;
  dst[idx] = (k < K && n < N) ? f2bf(src[(size_t)n * K + k]) : (unsigned short)0;
}

__global__ void k_copy_init(const float* __restrict__ gf, float* __restrict__ upd,
                            unsigned short* __restrict__ upd_bf, int n) {
  int i = blockIdx.x * 256 + threadIdx.x;
  if (i < n) { float v = gf[i]; upd[i] = v; upd_bf[i] = f2bf(v); }
}

// ---------------- h_t init + prev init ----------------
__global__ void k_init(const float* __restrict__ gf, float* h_t, unsigned short* ht_bf,
                       float* prev) {
  int b = blockIdx.x, tid = threadIdx.x;  // 256 threads
  __shared__ float red[8];
  __shared__ float wgt[Nm_];
  float imp = -3.4e38f;
  if (tid < Nm_) {
    const float* row = gf + (size_t)(b * Nm_ + tid) * Dm;
    float s = 0.f;
    for (int c = 0; c < Dm; c++) { float x = row[c]; s += x * x; }
    imp = sqrtf(s);
  }
  float M = blockMax<8>(imp, red);
  float e = (tid < Nm_) ? __expf(imp - M) : 0.f;
  float S = blockSum<8>(e, red);
  if (tid < Nm_) wgt[tid] = e / S;
  __syncthreads();
  for (int c = tid; c < Dm; c += 256) {
    float s = 0.f;
    for (int p = 0; p < Nm_; p++) s += gf[(size_t)(b * Nm_ + p) * Dm + c] * wgt[p];
    s += nrmf(0u, (unsigned)b, (unsigned)c) * 0.02f;
    h_t[b * Dm + c] = s; ht_bf[b * Dm + c] = f2bf(s);
  }
  if (tid < 2) {
    float ec = u01(hash3(1u, (unsigned)b, (unsigned)tid));
    float rr = u01(hash3(2u, (unsigned)b, (unsigned)tid));
    prev[b * 2 + tid] = (ec < 0.5f) ? rr * 0.2f : rr * 0.2f + 0.8f;
  }
}

// ---------------- patch bilinear sample -> im2col bf16 ----------------
__global__ void k_patch(const float* __restrict__ images, const float* __restrict__ prev,
                        unsigned short* __restrict__ X) {
  int tk = blockIdx.x;                       // 3136 token blocks
  int b = tk / Nm_, p = tk % Nm_;
  int py = p / FS, px = p % FS;
  float t00, t11, t02, t12;
  affineParams(prev, b, t00, t11, t02, t12);
  for (int k = threadIdx.x; k < 768; k += 256) {
    int c = k >> 8, ij = k & 255, i = ij >> 4, j = ij & 15;
    int u = px * 16 + j, v = py * 16 + i;
    float xs = ((float)u + 0.5f) * (2.f / (float)PS) - 1.f;
    float ys = ((float)v + 0.5f) * (2.f / (float)PS) - 1.f;
    float gx = t00 * xs + t02, gy = t11 * ys + t12;
    float ix = ((gx + 1.f) * (float)IMW - 1.f) * 0.5f;
    float iy = ((gy + 1.f) * (float)IMH - 1.f) * 0.5f;
    const float* img = images + (size_t)(b * 3 + c) * IMH * IMW;
    X[(size_t)tk * 768 + k] = f2bf(bilin(img, IMH, IMW, iy, ix));
  }
}

// ---------------- LayerNorm rows (in-place safe), optional gelu + bf16 mirror ----------------
__global__ void k_ln(const float* X, const float* g, const float* bta,
                     float* Y, unsigned short* Ybf, int Dlen, int doGelu) {
  int r = blockIdx.x;
  __shared__ float red[4];
  const float* x = X + (size_t)r * Dlen;
  float s = 0.f;
  for (int c = threadIdx.x; c < Dlen; c += 128) s += x[c];
  s = blockSum<4>(s, red);
  float mean = s / (float)Dlen;
  float v = 0.f;
  for (int c = threadIdx.x; c < Dlen; c += 128) { float d = x[c] - mean; v += d * d; }
  v = blockSum<4>(v, red);
  float rinv = rsqrtf(v / (float)Dlen + 1e-5f);
  for (int c = threadIdx.x; c < Dlen; c += 128) {
    float yv = (x[c] - mean) * rinv * g[c] + bta[c];
    if (doGelu) yv = geluf(yv);
    Y[(size_t)r * Dlen + c] = yv;
    if (Ybf) Ybf[(size_t)r * Dlen + c] = f2bf(yv);
  }
}

// ---------------- focus gate + feature grid-sample + upd update ----------------
__global__ void k_fuse(const float* __restrict__ fuh, const float* __restrict__ fu_w2,
                       const float* __restrict__ fu_b2, const float* __restrict__ local,
                       const float* __restrict__ prev,
                       float* __restrict__ upd, unsigned short* __restrict__ upd_bf) {
  int tk = blockIdx.x;
  int b = tk / Nm_, p = tk % Nm_;
  int py = p / FS, px = p % FS;
  int tid = threadIdx.x;  // 256
  __shared__ float red[8];
  __shared__ float fwS;
  float s = 0.f;
  for (int j = tid; j < DIm; j += 256) s += fuh[(size_t)tk * DIm + j] * fu_w2[j];
  s = blockSum<8>(s, red);
  if (tid == 0) fwS = sigf(s + fu_b2[0]);
  float t00, t11, t02, t12;
  affineParams(prev, b, t00, t11, t02, t12);
  float xs = ((float)px + 0.5f) * (2.f / (float)FS) - 1.f;
  float ys = ((float)py + 0.5f) * (2.f / (float)FS) - 1.f;
  float gx = t00 * xs + t02, gy = t11 * ys + t12;
  float ix = ((gx + 1.f) * (float)FS - 1.f) * 0.5f;
  float iy = ((gy + 1.f) * (float)FS - 1.f) * 0.5f;
  float fx = floorf(ix), fy = floorf(iy);
  int x0 = (int)fx, y0 = (int)fy;
  float wx1 = ix - fx, wy1 = iy - fy;
  __syncthreads();
  float f = fwS;
  for (int c = tid; c < Dm; c += 256) {
    float sm = 0.f;
    #pragma unroll
    for (int dy = 0; dy < 2; dy++)
      #pragma unroll
      for (int dx = 0; dx < 2; dx++) {
        int xx = x0 + dx, yy = y0 + dy;
        if (xx >= 0 && xx < FS && yy >= 0 && yy < FS) {
          float w = (dx ? wx1 : 1.f - wx1) * (dy ? wy1 : 1.f - wy1);
          sm += local[(size_t)(b * Nm_ + yy * FS + xx) * Dm + c] * w;
        }
      }
    float nu = upd[(size_t)tk * Dm + c] + f * sm;
    upd[(size_t)tk * Dm + c] = nu;
    upd_bf[(size_t)tk * Dm + c] = f2bf(nu);
  }
}

// ---------------- per-batch token means of local & upd ----------------
__global__ void k_means(const float* local, const float* upd,
                        float* am, float* bm, unsigned short* ab_bf) {
  int b = blockIdx.x, tid = threadIdx.x;  // 256
  for (int c = tid; c < Dm; c += 256) {
    float sa = 0.f, sb = 0.f;
    for (int p = 0; p < Nm_; p++) {
      sa += local[(size_t)(b * Nm_ + p) * Dm + c];
      sb += upd[(size_t)(b * Nm_ + p) * Dm + c];
    }
    sa *= (1.f / (float)Nm_); sb *= (1.f / (float)Nm_);
    am[b * Dm + c] = sa; bm[b * Dm + c] = sb;
    ab_bf[b * 2 * Dm + c] = f2bf(sa);
    ab_bf[b * 2 * Dm + Dm + c] = f2bf(sb);
  }
}

__global__ void k_fusedf(const float* gate, const float* am, const float* bm, float* fusedf) {
  int i = blockIdx.x * 256 + threadIdx.x;
  if (i >= Bm_ * Dm) return;
  float g = gate[i];
  fusedf[i] = g * am[i] + (1.f - g) * bm[i];
}

// ---------------- positional encoder MLP + LN ----------------
__global__ void k_pe(const float* prev, const float* w1, const float* b1,
                     const float* w2, const float* b2,
                     const float* g, const float* bet, float* pe_out) {
  int b = blockIdx.x, tid = threadIdx.x;  // 128
  __shared__ float h1[96];
  __shared__ float pv[Dm];
  __shared__ float red[4];
  float p0 = prev[b * 2], p1 = prev[b * 2 + 1];
  if (tid < 96) h1[tid] = geluf(w1[tid * 2] * p0 + w1[tid * 2 + 1] * p1 + b1[tid]);
  __syncthreads();
  for (int c = tid; c < Dm; c += 128) {
    float s = b2[c];
    for (int j = 0; j < 96; j++) s += w2[c * 96 + j] * h1[j];
    pv[c] = s;
  }
  __syncthreads();
  float s = 0.f;
  for (int c = tid; c < Dm; c += 128) s += pv[c];
  s = blockSum<4>(s, red);
  float mean = s / (float)Dm;
  float v = 0.f;
  for (int c = tid; c < Dm; c += 128) { float d = pv[c] - mean; v += d * d; }
  v = blockSum<4>(v, red);
  float rinv = rsqrtf(v / (float)Dm + 1e-5f);
  for (int c = tid; c < Dm; c += 128)
    pe_out[b * Dm + c] = (pv[c] - mean) * rinv * g[c] + bet[c];
}

// ---------------- single-query attention per (b,h) ----------------
__global__ void k_attn(const float* __restrict__ qb, const float* __restrict__ kmat,
                       const float* __restrict__ vmat,
                       float* __restrict__ attn_f, unsigned short* __restrict__ attn_bf) {
  int bh = blockIdx.x;
  int b = bh >> 3, h = bh & 7;
  int tid = threadIdx.x;  // 128
  __shared__ float sc[Nm_];
  __shared__ float red[4];
  __shared__ float qv[HDm];
  if (tid < HDm) qv[tid] = qb[b * Dm + h * HDm + tid];
  __syncthreads();
  for (int p = tid; p < Nm_; p += 128) {
    const float* kr = kmat + (size_t)(b * Nm_ + p) * Dm + h * HDm;
    float s = 0.f;
    for (int d = 0; d < HDm; d++) s += qv[d] * kr[d];
    sc[p] = s * 0.14433756729f;  // 1/sqrt(48)
  }
  __syncthreads();
  float m = -3.4e38f;
  for (int p = tid; p < Nm_; p += 128) m = fmaxf(m, sc[p]);
  m = blockMax<4>(m, red);
  float s = 0.f;
  for (int p = tid; p < Nm_; p += 128) { float e = __expf(sc[p] - m); sc[p] = e; s += e; }
  s = blockSum<4>(s, red);
  float inv = 1.f / s;
  if (tid < HDm) {
    float o = 0.f;
    for (int p = 0; p < Nm_; p++) o += sc[p] * vmat[(size_t)(b * Nm_ + p) * Dm + h * HDm + tid];
    o *= inv;
    attn_f[b * Dm + h * HDm + tid] = o;
    attn_bf[b * Dm + h * HDm + tid] = f2bf(o);
  }
}

__global__ void k_concat(const float* h_t, const float* attn, const float* fusedf,
                         const float* pe, unsigned short* dst) {
  int i = blockIdx.x * 256 + threadIdx.x;
  if (i >= Bm_ * 4 * Dm) return;
  int b = i / (4 * Dm), c = i % (4 * Dm);
  float v = (c < Dm)       ? h_t[b * Dm + c]
          : (c < 2 * Dm)   ? attn[b * Dm + c - Dm]
          : (c < 3 * Dm)   ? fusedf[b * Dm + c - 2 * Dm]
                           : pe[b * Dm + c - 3 * Dm];
  dst[i] = f2bf(v);
}

// ---------------- mamba pointwise stages ----------------
__global__ void k_mamba_a(const float* xz, const float* mconv, const float* mconvb,
                          float* xm, unsigned short* xm_bf) {
  int i = blockIdx.x * 256 + threadIdx.x;
  if (i >= Bm_ * DIm) return;
  int b = i / DIm, c = i % DIm;
  float x = xz[b * 2 * DIm + c] * mconv[c * 4 + 3] + mconvb[c];
  float v = x * sigf(x);  // silu
  xm[i] = v; xm_bf[i] = f2bf(v);
}

__global__ void k_mamba_b(const float* xdbl, float* bc, unsigned short* dtin) {
  int b = blockIdx.x, tid = threadIdx.x;  // 256
  __shared__ float red[8];
  float s = xdbl[b * 536 + DTR + tid] * xdbl[b * 536 + DTR + DSm + tid];
  s = blockSum<8>(s, red);
  if (tid == 0) bc[b] = s;
  if (tid < 32) dtin[b * 32 + tid] = (tid < DTR) ? f2bf(xdbl[b * 536 + tid]) : (unsigned short)0;
}

__global__ void k_mamba_c(const float* dt_pre, const float* xm, const float* xz,
                          const float* mD, const float* bc, unsigned short* y_bf) {
  int i = blockIdx.x * 256 + threadIdx.x;
  if (i >= Bm_ * DIm) return;
  int b = i / DIm, c = i % DIm;
  float d = dt_pre[i];
  float sp = (d > 20.f) ? d : log1pf(__expf(d));  // softplus
  float z = xz[b * 2 * DIm + DIm + c];
  float y = (sp * xm[i] * bc[b] + mD[c] * xm[i]) * (z * sigf(z));
  y_bf[i] = f2bf(y);
}

// ---------------- stop prob + comb ----------------
__global__ void k_stop_comb(const float* scv, const float* w2, const float* b2,
                            const float* h_t, const float* fusedf,
                            float* stops_out, unsigned short* comb_bf, int t) {
  int b = blockIdx.x, tid = threadIdx.x;  // 128
  __shared__ float red[4];
  float s = 0.f;
  for (int c = tid; c < 192; c += 128) s += scv[b * 192 + c] * w2[c];
  s = blockSum<4>(s, red);
  if (tid == 0) stops_out[b * Tm_ + t] = 1.f - sigf(s + b2[0]);
  for (int c = tid; c < Dm; c += 128)
    comb_bf[b * Dm + c] = f2bf(h_t[b * Dm + c] + fusedf[b * Dm + c]);
}

// ---------------- VAE reparameterize + output mus/logvars ----------------
__global__ void k_vae(const float* mu, const float* lv, float* mus_out, float* lvs_out,
                      unsigned short* zlat_bf, int t) {
  int b = blockIdx.x, c = threadIdx.x;  // 192 threads
  float m = mu[b * 192 + c], l = lv[b * 192 + c];
  mus_out[(size_t)(b * Tm_ + t) * 192 + c] = m;
  lvs_out[(size_t)(b * Tm_ + t) * 192 + c] = l;
  float eps = nrmf(100u + (unsigned)t, (unsigned)b, (unsigned)c);
  zlat_bf[b * 192 + c] = f2bf(m + eps * __expf(0.5f * l));
}

// ---------------- position decode ----------------
__global__ void k_pos(const float* pdv, const float* w2, const float* b2,
                      float* pos_out, float* prev, int t) {
  int b = blockIdx.x, tid = threadIdx.x;  // 128
  __shared__ float red[4];
  float v = (tid < 96) ? pdv[b * 96 + tid] : 0.f;
  float s0 = blockSum<4>((tid < 96) ? v * w2[tid] : 0.f, red);
  float s1 = blockSum<4>((tid < 96) ? v * w2[96 + tid] : 0.f, red);
  if (tid < 2) {
    float s = (tid == 0) ? s0 : s1;
    float pos = tanhf(s + b2[tid]);
    pos = fminf(fmaxf((pos + 1.f) * 0.5f, 0.f), 1.f);
    pos_out[(size_t)(b * Tm_ + t) * 2 + tid] = pos;
    prev[b * 2 + tid] = pos;
  }
}

__global__ void k_finish(const float* upd, float* out_upd, int n, int* out_actual) {
  int i = blockIdx.x * 256 + threadIdx.x;
  if (i < n) out_upd[i] = upd[i];
  if (i < Bm_) out_actual[i] = Tm_;
}

// =============================================================================
extern "C" void kernel_launch(void* const* d_in, const int* in_sizes, int n_in,
                              void* d_out, int out_size, void* d_ws, size_t ws_size,
                              hipStream_t stream) {
  (void)in_sizes; (void)n_in; (void)out_size; (void)ws_size;
  const float* images = (const float*)d_in[0];
  const float* gfeat  = (const float*)d_in[1];
  // d_in[2] = seq_len (==8, hardcoded)
  // params: sorted-key pytree order
  #define P(i) ((const float*)d_in[3 + (i)])
  // 0 bk 1 bo 2 bq 3 bv 4 ff_b 5 ff_ln_b 6 ff_ln_g 7 ff_w 8 focus_b 9 focus_ln_b
  // 10 focus_ln_g 11 focus_w 12 fu_b1 13 fu_b2 14 fu_w1 15 fu_w2 16 gf_b 17 gf_w
  // 18 lv_b 19 lv_w 20 m_D 21 m_conv 22 m_conv_b 23 m_dt 24 m_dt_b 25 m_in 26 m_out
  // 27 m_xproj 28 mu_b 29 mu_w 30 pd_b1 31 pd_b2 32 pd_ln_b 33 pd_ln_g 34 pd_w1
  // 35 pd_w2 36 pe_b1 37 pe_b2 38 pe_ln_b 39 pe_ln_g 40 pe_w1 41 pe_w2 42 sc_b1
  // 43 sc_b2 44 sc_ln_b 45 sc_ln_g 46 sc_w1 47 sc_w2 48 wk 49 wo 50 wq 51 wv

  float* outF = (float*)d_out;
  const size_t OFF_POS = 0;
  const size_t OFF_UPD = OFF_POS + (size_t)Bm_ * Tm_ * 2;
  const size_t OFF_MUS = OFF_UPD + (size_t)Bm_ * Nm_ * Dm;
  const size_t OFF_LVS = OFF_MUS + (size_t)Bm_ * Tm_ * 192;
  const size_t OFF_STP = OFF_LVS + (size_t)Bm_ * Tm_ * 192;
  const size_t OFF_ACT = OFF_STP + (size_t)Bm_ * Tm_;

  // ---- workspace bump allocator ----
  char* wsp = (char*)d_ws; size_t off = 0;
  auto alloc = [&](size_t bytes) -> void* {
    void* p = wsp + off; off = (off + bytes + 255) & ~(size_t)255; return p;
  };
  const int TOK = Bm_ * Nm_;  // 3136
  float* h_t     = (float*)alloc(Bm_ * Dm * 4);
  float* prev    = (float*)alloc(Bm_ * 2 * 4);
  float* upd     = (float*)alloc((size_t)TOK * Dm * 4);
  float* local   = (float*)alloc((size_t)TOK * Dm * 4);
  float* fuh     = (float*)alloc((size_t)TOK * DIm * 4);
  float* kmat    = (float*)alloc((size_t)TOK * Dm * 4);
  float* vmat    = (float*)alloc((size_t)TOK * Dm * 4);
  float* qbuf    = (float*)alloc(Bm_ * Dm * 4);
  float* attn_f  = (float*)alloc(Bm_ * Dm * 4);
  float* attn_o  = (float*)alloc(Bm_ * Dm * 4);
  float* am      = (float*)alloc(Bm_ * Dm * 4);
  float* bmn     = (float*)alloc(Bm_ * Dm * 4);
  float* gate    = (float*)alloc(Bm_ * Dm * 4);
  float* fusedf  = (float*)alloc(Bm_ * Dm * 4);
  float* pe_out  = (float*)alloc(Bm_ * Dm * 4);
  float* ff_pre  = (float*)alloc(Bm_ * Dm * 4);
  float* fusedv  = (float*)alloc(Bm_ * Dm * 4);
  float* xz      = (float*)alloc(Bm_ * 2 * DIm * 4);
  float* xm      = (float*)alloc(Bm_ * DIm * 4);
  float* xdbl    = (float*)alloc(Bm_ * 536 * 4);
  float* dt_pre  = (float*)alloc(Bm_ * DIm * 4);
  float* bc      = (float*)alloc(Bm_ * 4);
  float* sc_pre  = (float*)alloc(Bm_ * 192 * 4);
  float* scv     = (float*)alloc(Bm_ * 192 * 4);
  float* muv     = (float*)alloc(Bm_ * 192 * 4);
  float* lvv     = (float*)alloc(Bm_ * 192 * 4);
  float* pd_pre  = (float*)alloc(Bm_ * 96 * 4);
  float* pdv     = (float*)alloc(Bm_ * 96 * 4);

  unsigned short* ht_bf    = (unsigned short*)alloc(Bm_ * Dm * 2);
  unsigned short* upd_bf   = (unsigned short*)alloc((size_t)TOK * Dm * 2);
  unsigned short* Xim      = (unsigned short*)alloc((size_t)TOK * 768 * 2);
  unsigned short* local_bf = (unsigned short*)alloc((size_t)TOK * Dm * 2);
  unsigned short* ab_bf    = (unsigned short*)alloc(Bm_ * 2 * Dm * 2);
  unsigned short* attn_bf  = (unsigned short*)alloc(Bm_ * Dm * 2);
  unsigned short* fin_bf   = (unsigned short*)alloc(Bm_ * 4 * Dm * 2);
  unsigned short* fused_bf = (unsigned short*)alloc(Bm_ * Dm * 2);
  unsigned short* xm_bf    = (unsigned short*)alloc(Bm_ * DIm * 2);
  unsigned short* dtin_bf  = (unsigned short*)alloc(Bm_ * 32 * 2);
  unsigned short* y_bf     = (unsigned short*)alloc(Bm_ * DIm * 2);
  unsigned short* comb_bf  = (unsigned short*)alloc(Bm_ * Dm * 2);
  unsigned short* zlat_bf  = (unsigned short*)alloc(Bm_ * 192 * 2);

  // packed weights: Kpad * Npad bf16 each (Npad = ceil(N/64)*64)
  unsigned short* Wfocus = (unsigned short*)alloc((size_t)768 * 384 * 2);
  unsigned short* Wfu1   = (unsigned short*)alloc((size_t)384 * 768 * 2);
  unsigned short* Wgf    = (unsigned short*)alloc((size_t)768 * 384 * 2);
  unsigned short* Wq     = (unsigned short*)alloc((size_t)384 * 384 * 2);
  unsigned short* Wk     = (unsigned short*)alloc((size_t)384 * 384 * 2);
  unsigned short* Wv     = (unsigned short*)alloc((size_t)384 * 384 * 2);
  unsigned short* Wo     = (unsigned short*)alloc((size_t)384 * 384 * 2);
  unsigned short* Wff    = (unsigned short*)alloc((size_t)1536 * 384 * 2);
  unsigned short* Wmin   = (unsigned short*)alloc((size_t)384 * 1536 * 2);
  unsigned short* Wxp    = (unsigned short*)alloc((size_t)768 * 576 * 2);
  unsigned short* Wdt    = (unsigned short*)alloc((size_t)32 * 768 * 2);
  unsigned short* Wmout  = (unsigned short*)alloc((size_t)768 * 384 * 2);
  unsigned short* Wsc1   = (unsigned short*)alloc((size_t)384 * 192 * 2);
  unsigned short* Wmu    = (unsigned short*)alloc((size_t)384 * 192 * 2);
  unsigned short* Wlv    = (unsigned short*)alloc((size_t)384 * 192 * 2);
  unsigned short* Wpd1   = (unsigned short*)alloc((size_t)192 * 128 * 2);

  auto wprep = [&](const float* src, unsigned short* dst, int N, int K, int Kpad) {
    int Npad = ((N + 63) / 64) * 64;
    int n = Kpad * Npad;
    k_wprep<<<(n + 255) / 256, 256, 0, stream>>>(src, dst, N, K, Kpad, Npad);
  };
  wprep(P(11), Wfocus, 384, 768, 768);
  wprep(P(14), Wfu1, 768, 384, 384);
  wprep(P(17), Wgf, 384, 768, 768);
  wprep(P(50), Wq, 384, 384, 384);
  wprep(P(48), Wk, 384, 384, 384);
  wprep(P(51), Wv, 384, 384, 384);
  wprep(P(49), Wo, 384, 384, 384);
  wprep(P(7),  Wff, 384, 1536, 1536);
  wprep(P(25), Wmin, 1536, 384, 384);
  wprep(P(27), Wxp, 536, 768, 768);
  wprep(P(23), Wdt, 768, 24, 32);
  wprep(P(26), Wmout, 384, 768, 768);
  wprep(P(46), Wsc1, 192, 384, 384);
  wprep(P(29), Wmu, 192, 384, 384);
  wprep(P(19), Wlv, 192, 384, 384);
  wprep(P(34), Wpd1, 96, 192, 192);

  auto gemm = [&](const unsigned short* A, int lda, const unsigned short* Bp,
                  const float* bias, float* C, unsigned short* Cbf, int ldc,
                  int M, int N, int K, int act) {
    dim3 g((N + 63) / 64, (M + 31) / 32);
    if (act == 1)      k_gemm<1><<<g, 128, 0, stream>>>(A, lda, Bp, bias, C, Cbf, ldc, M, N, K);
    else if (act == 2) k_gemm<2><<<g, 128, 0, stream>>>(A, lda, Bp, bias, C, Cbf, ldc, M, N, K);
    else               k_gemm<0><<<g, 128, 0, stream>>>(A, lda, Bp, bias, C, Cbf, ldc, M, N, K);
  };

  // init
  k_copy_init<<<((TOK * Dm) + 255) / 256, 256, 0, stream>>>(gfeat, upd, upd_bf, TOK * Dm);
  k_init<<<Bm_, 256, 0, stream>>>(gfeat, h_t, ht_bf, prev);

  for (int t = 0; t < Tm_; ++t) {
    // patch embed (im2col + WMMA GEMM) + LN
    k_patch<<<TOK, 256, 0, stream>>>(images, prev, Xim);
    gemm(Xim, 768, Wfocus, P(8), local, nullptr, Dm, TOK, Dm, 768, 0);
    k_ln<<<TOK, 128, 0, stream>>>(local, P(10), P(9), local, local_bf, Dm, 0);
    // focus gate MLP hidden
    gemm(local_bf, Dm, Wfu1, P(12), fuh, nullptr, DIm, TOK, DIm, Dm, 1);
    // gate + grid-sample + upd update
    k_fuse<<<TOK, 256, 0, stream>>>(fuh, P(15), P(13), local, prev, upd, upd_bf);
    // means + fusion gate
    k_means<<<Bm_, 256, 0, stream>>>(local, upd, am, bmn, ab_bf);
    gemm(ab_bf, 2 * Dm, Wgf, P(16), gate, nullptr, Dm, Bm_, Dm, 2 * Dm, 2);
    k_fusedf<<<(Bm_ * Dm + 255) / 256, 256, 0, stream>>>(gate, am, bmn, fusedf);
    // positional encoder
    k_pe<<<Bm_, 128, 0, stream>>>(prev, P(40), P(36), P(41), P(37), P(39), P(38), pe_out);
    // attention: q (old h_t), k, v on updated tokens
    gemm(ht_bf, Dm, Wq, P(2), qbuf, nullptr, Dm, Bm_, Dm, Dm, 0);
    gemm(upd_bf, Dm, Wk, P(0), kmat, nullptr, Dm, TOK, Dm, Dm, 0);
    gemm(upd_bf, Dm, Wv, P(3), vmat, nullptr, Dm, TOK, Dm, Dm, 0);
    k_attn<<<Bm_ * NHm, 128, 0, stream>>>(qbuf, kmat, vmat, attn_f, attn_bf);
    gemm(attn_bf, Dm, Wo, P(1), attn_o, nullptr, Dm, Bm_, Dm, Dm, 0);
    // fuse + LN + gelu
    k_concat<<<(Bm_ * 4 * Dm + 255) / 256, 256, 0, stream>>>(h_t, attn_o, fusedf, pe_out, fin_bf);
    gemm(fin_bf, 4 * Dm, Wff, P(4), ff_pre, nullptr, Dm, Bm_, Dm, 4 * Dm, 0);
    k_ln<<<Bm_, 128, 0, stream>>>(ff_pre, P(6), P(5), fusedv, fused_bf, Dm, 1);
    // mamba-ish block (scan collapsed to dt*xm*(Bm.Cm))
    gemm(fused_bf, Dm, Wmin, nullptr, xz, nullptr, 2 * DIm, Bm_, 2 * DIm, Dm, 0);
    k_mamba_a<<<(Bm_ * DIm + 255) / 256, 256, 0, stream>>>(xz, P(21), P(22), xm, xm_bf);
    gemm(xm_bf, DIm, Wxp, nullptr, xdbl, nullptr, 536, Bm_, 536, DIm, 0);
    k_mamba_b<<<Bm_, 256, 0, stream>>>(xdbl, bc, dtin_bf);
    gemm(dtin_bf, 32, Wdt, P(24), dt_pre, nullptr, DIm, Bm_, DIm, 32, 0);
    k_mamba_c<<<(Bm_ * DIm + 255) / 256, 256, 0, stream>>>(dt_pre, xm, xz, P(20), bc, y_bf);
    gemm(y_bf, DIm, Wmout, nullptr, h_t, ht_bf, Dm, Bm_, Dm, DIm, 0);
    // stop head
    gemm(ht_bf, Dm, Wsc1, P(42), sc_pre, nullptr, 192, Bm_, 192, Dm, 0);
    k_ln<<<Bm_, 128, 0, stream>>>(sc_pre, P(45), P(44), scv, nullptr, 192, 1);
    k_stop_comb<<<Bm_, 128, 0, stream>>>(scv, P(47), P(43), h_t, fusedf,
                                         outF + OFF_STP, comb_bf, t);
    // VAE heads
    gemm(comb_bf, Dm, Wmu, P(28), muv, nullptr, 192, Bm_, 192, Dm, 0);
    gemm(comb_bf, Dm, Wlv, P(18), lvv, nullptr, 192, Bm_, 192, Dm, 0);
    k_vae<<<Bm_, 192, 0, stream>>>(muv, lvv, outF + OFF_MUS, outF + OFF_LVS, zlat_bf, t);
    // position decoder
    gemm(zlat_bf, 192, Wpd1, P(30), pd_pre, nullptr, 96, Bm_, 96, 192, 0);
    k_ln<<<Bm_, 128, 0, stream>>>(pd_pre, P(33), P(32), pdv, nullptr, 96, 1);
    k_pos<<<Bm_, 128, 0, stream>>>(pdv, P(35), P(31), outF + OFF_POS, prev, t);
  }

  int nUpd = TOK * Dm;
  k_finish<<<(nUpd + 255) / 256, 256, 0, stream>>>(upd, outF + OFF_UPD, nUpd,
                                                   (int*)(outF + OFF_ACT));
  #undef P
}
